// Engram_21466246546079
// MI455X (gfx1250) — compile-verified, matrix operand order checked by reading, not run on hardware
//
#include <hip/hip_runtime.h>
#include <hip/hip_bf16.h>

typedef __attribute__((ext_vector_type(16))) __bf16 v16bf;
typedef __attribute__((ext_vector_type(8)))  float  v8f;
typedef int v4i_vs __attribute__((vector_size(16)));

#define B_   2
#define L_   4096
#define HID_ 2048
#define ENG_ 1024
#define NH_  16
#define DH_  64
#define VOCAB_ 129280
#define HC_  4
#define KSZ_ 4
#define DIL_ 3
#define M_   (B_ * L_)          // 8192 rows
#define N_   ((HC_ + 1) * HID_) // 10240 output channels (4 key heads + value)
#define K_   ENG_               // 1024
#define EPS_RMS 1.1920929e-07f
#define EPS_SC  1e-05f

// ---- CDNA5 async global->LDS copy (guarded; falls back to reg staging) ----
#if __has_builtin(__builtin_amdgcn_global_load_async_to_lds_b128)
#define HAVE_ASYNC_LDS 1
typedef __attribute__((address_space(1))) v4i_vs* as1_v4i_p;
typedef __attribute__((address_space(3))) v4i_vs* as3_v4i_p;
#else
#define HAVE_ASYNC_LDS 0
#endif

__device__ __forceinline__ void copy16_g2l(__bf16* dst_lds, const __bf16* src_g) {
#if HAVE_ASYNC_LDS
  // param1: v4i addrspace(1)* (global src); param2: LDS dst.
  // Integer round-trips perform the address-space conversions C++ casts refuse;
  // low 32 bits of a generic LDS pointer are the LDS byte address (ISA 10.2).
  as1_v4i_p g = (as1_v4i_p)(unsigned long long)src_g;
  as3_v4i_p l = (as3_v4i_p)(unsigned int)(unsigned long long)dst_lds;
  __builtin_amdgcn_global_load_async_to_lds_b128(g, l, 0, 0);
#else
  *(uint4*)dst_lds = *(const uint4*)src_g;
#endif
}

__device__ __forceinline__ void wait_async_lds() {
#if HAVE_ASYNC_LDS
#if __has_builtin(__builtin_amdgcn_s_wait_asynccnt)
  __builtin_amdgcn_s_wait_asynccnt(0);
#else
  asm volatile("s_wait_asynccnt 0" ::: "memory");
#endif
#endif
}

// ---------------- 1) hash gather -> bf16 A matrix [M_, K_] ----------------
__global__ __launch_bounds__(256) void gather_k(
    const float* __restrict__ tabs, const long long* __restrict__ ids,
    __bf16* __restrict__ embB) {
  size_t idx = (size_t)blockIdx.x * 256 + threadIdx.x;  // over M_*K_ = 8Mi
  int d  = (int)(idx & (DH_ - 1));
  int hh = (int)((idx >> 6) & (NH_ - 1));
  size_t bl = idx >> 10;
  long long id = ids[bl * NH_ + hh];
  float v = tabs[((size_t)hh * VOCAB_ + (size_t)id) * DH_ + d];
  embB[bl * ENG_ + (size_t)hh * DH_ + d] = (__bf16)v;
}

// ------- 2) weight transpose+cvt -> bf16 Bt [N_, K_] (row-major, n-major) -------
__global__ __launch_bounds__(256) void transpose_k(
    const float* __restrict__ key_w, const float* __restrict__ value_w,
    __bf16* __restrict__ wt) {
  __shared__ float tile[32][33];
  int g  = blockIdx.z;                 // 0..4 (4 = value_w)
  int h0 = blockIdx.x * 32;            // output-channel (HID) tile
  int e0 = blockIdx.y * 32;            // K (ENG) tile
  const float* src = (g < HC_) ? key_w + (size_t)g * ENG_ * HID_ : value_w;
  int tx = threadIdx.x, ty = threadIdx.y;   // (32, 8)
  for (int i = 0; i < 32; i += 8)
    tile[ty + i][tx] = src[(size_t)(e0 + ty + i) * HID_ + h0 + tx];
  __syncthreads();
  for (int i = 0; i < 32; i += 8)
    wt[((size_t)g * HID_ + h0 + ty + i) * (size_t)K_ + e0 + tx] =
        (__bf16)tile[tx][ty + i];
}

// ------- 3) bf16 WMMA GEMM: proj[M_, N_] = A x Bt^T, 128x256 block tile -------
__global__ __launch_bounds__(256) void gemm_bf16(
    const __bf16* __restrict__ A, const __bf16* __restrict__ Bt,
    float* __restrict__ C) {
  __shared__ __bf16 As[2][128 * 32];   // 16 KiB
  __shared__ __bf16 Bs[2][256 * 32];   // 32 KiB
  const int tid  = threadIdx.x;
  const int lane = tid & 31;
  const int wave = tid >> 5;
  const int wm = wave >> 2;   // 0..1 -> 64 rows each
  const int wn = wave & 3;    // 0..3 -> 64 cols each
  const int rowA0 = blockIdx.y * 128;   // M tile base
  const int rowB0 = blockIdx.x * 256;   // N tile base

  v8f acc[4][4];
  for (int i = 0; i < 4; ++i)
    for (int j = 0; j < 4; ++j)
      for (int e = 0; e < 8; ++e) acc[i][j][e] = 0.0f;

  auto load_tile = [&](int buf, int k0) {
    // A: 128x32 bf16 = 512 16B chunks (2/thread); B: 256x32 = 1024 (4/thread)
    for (int it = 0; it < 2; ++it) {
      int c = tid + it * 256;
      int r = c >> 2, q = c & 3;
      copy16_g2l(&As[buf][r * 32 + q * 8],
                 A + (size_t)(rowA0 + r) * K_ + k0 + q * 8);
    }
    for (int it = 0; it < 4; ++it) {
      int c = tid + it * 256;
      int r = c >> 2, q = c & 3;
      copy16_g2l(&Bs[buf][r * 32 + q * 8],
                 Bt + (size_t)(rowB0 + r) * K_ + k0 + q * 8);
    }
  };
  // 16-bit A/B fragment (16x32): lanes 0-15 -> K kh..kh+7 & kh+16..kh+23 with
  // kh=0; lanes 16-31 same with kh=8 (ISA 16-bit A/B-matrix layout).
  auto read_frag = [&](const __bf16* base, int row) -> v16bf {
    int m  = row + (lane & 15);
    int kh = (lane >> 4) << 3;
    union { uint4 u[2]; v16bf v; } f;
    f.u[0] = *(const uint4*)(base + m * 32 + kh);
    f.u[1] = *(const uint4*)(base + m * 32 + kh + 16);
    return f.v;
  };

  load_tile(0, 0);
  wait_async_lds();
  __syncthreads();
  const int KT = K_ / 32;
  for (int kt = 0; kt < KT; ++kt) {
    int buf = kt & 1;
    if (kt + 1 < KT) load_tile(buf ^ 1, (kt + 1) * 32);
    v16bf af[4], bf[4];
    for (int i = 0; i < 4; ++i) af[i] = read_frag(As[buf], wm * 64 + i * 16);
    for (int j = 0; j < 4; ++j) bf[j] = read_frag(Bs[buf], wn * 64 + j * 16);
    for (int i = 0; i < 4; ++i)
      for (int j = 0; j < 4; ++j)
        acc[i][j] = __builtin_amdgcn_wmma_f32_16x16x32_bf16(
            false, af[i], false, bf[j], (short)0, acc[i][j], false, false);
    wait_async_lds();
    __syncthreads();
  }

  // C/D layout: VGPR r, lanes 0-15 -> M=r, N=lane; lanes 16-31 -> M=r+8
  for (int i = 0; i < 4; ++i)
    for (int j = 0; j < 4; ++j) {
      int r0 = rowA0 + wm * 64 + i * 16 + ((lane >> 4) << 3);
      int c0 = rowB0 + wn * 64 + j * 16 + (lane & 15);
      for (int r = 0; r < 8; ++r)
        C[(size_t)(r0 + r) * N_ + c0] = acc[i][j][r];
    }
}

// ---------------- 4) gate + conv-RMS stats per (row, g) ----------------
__global__ __launch_bounds__(256) void gate_k(
    const float* __restrict__ proj, const float* __restrict__ hs,
    const float* __restrict__ key_b, const float* __restrict__ value_b,
    const float* __restrict__ n1w, const float* __restrict__ n2w,
    float* __restrict__ gate_out, float* __restrict__ rstdv_out) {
  int row = blockIdx.x;   // 0..M_-1
  int g   = blockIdx.y;   // 0..HC_-1
  const float* pk = proj + (size_t)row * N_ + (size_t)g * HID_;
  const float* pv = proj + (size_t)row * N_ + (size_t)HC_ * HID_;
  const float* q  = hs   + (size_t)row * HID_;
  float sk = 0.f, sq = 0.f, skq = 0.f, sv = 0.f;
  for (int h = threadIdx.x; h < HID_; h += 256) {
    float k  = pk[h] + key_b[g * HID_ + h];
    float qv = q[h];
    float v  = pv[h] + value_b[h];
    sk  += k * k;
    sq  += qv * qv;
    skq += k * qv * n1w[g * HID_ + h] * n2w[g * HID_ + h];
    sv  += v * v;
  }
  __shared__ float red[8][4];
  int lane = threadIdx.x & 31, wave = threadIdx.x >> 5;
  for (int off = 16; off; off >>= 1) {
    sk  += __shfl_down(sk, off, 32);
    sq  += __shfl_down(sq, off, 32);
    skq += __shfl_down(skq, off, 32);
    sv  += __shfl_down(sv, off, 32);
  }
  if (lane == 0) { red[wave][0]=sk; red[wave][1]=sq; red[wave][2]=skq; red[wave][3]=sv; }
  __syncthreads();
  if (threadIdx.x == 0) {
    sk = sq = skq = sv = 0.f;
    for (int w = 0; w < 8; ++w) { sk+=red[w][0]; sq+=red[w][1]; skq+=red[w][2]; sv+=red[w][3]; }
    float rk = rsqrtf(sk / HID_ + EPS_RMS);
    float rq = rsqrtf(sq / HID_ + EPS_RMS);
    float graw = rk * rq * skq * (1.0f / sqrtf((float)HID_));
    float gs   = copysignf(sqrtf(fmaxf(fabsf(graw), 1e-6f)), graw);
    float gate = 1.0f / (1.0f + __expf(-gs));
    gate_out[row * HC_ + g]  = gate;
    rstdv_out[row * HC_ + g] = rsqrtf(gate * gate * sv / HID_ + EPS_SC);
  }
}

// ------- 5) gated value + dilated depthwise conv + SiLU + head-sum -------
__global__ __launch_bounds__(256) void out_k(
    const float* __restrict__ proj, const float* __restrict__ value_b,
    const float* __restrict__ gate, const float* __restrict__ rstdv,
    const float* __restrict__ scw, const float* __restrict__ cw,
    float* __restrict__ out) {
  size_t idx = (size_t)blockIdx.x * 256 + threadIdx.x;  // over M_*HID_
  int h = (int)(idx & (HID_ - 1));
  size_t bl = idx >> 11;
  int l = (int)(bl & (L_ - 1));
  size_t b = bl >> 12;
  float vb = value_b[h];
  float pvb[KSZ_];
  int tt[KSZ_];
  for (int k = 0; k < KSZ_; ++k) {
    int t = l - (KSZ_ - 1) * DIL_ + k * DIL_;   // l - 9 + 3k
    tt[k] = t;
    pvb[k] = (t >= 0)
        ? proj[(b * L_ + (size_t)t) * N_ + (size_t)HC_ * HID_ + h] + vb
        : 0.0f;
  }
  float acc = 0.0f;
  for (int g = 0; g < HC_; ++g) {
    float w  = scw[g * HID_ + h];
    float conv = 0.0f;
    for (int k = 0; k < KSZ_; ++k) {
      if (tt[k] >= 0) {
        size_t tg = (b * L_ + (size_t)tt[k]) * HC_ + g;
        conv += cw[(size_t)(g * HID_ + h) * KSZ_ + k] *
                (pvb[k] * gate[tg]) * rstdv[tg] * w;
      }
    }
    float val_l = pvb[KSZ_ - 1] * gate[bl * HC_ + g];   // t == l
    acc += val_l + conv / (1.0f + __expf(-conv));       // + silu(conv)
  }
  out[idx] = acc;
}

extern "C" void kernel_launch(void* const* d_in, const int* in_sizes, int n_in,
                              void* d_out, int out_size, void* d_ws, size_t ws_size,
                              hipStream_t stream) {
  (void)in_sizes; (void)n_in; (void)out_size; (void)ws_size;
  const float*     hidden  = (const float*)d_in[0];
  const long long* hashids = (const long long*)d_in[1];   // jnp.int64
  const float*     tabs    = (const float*)d_in[2];
  const float*     value_w = (const float*)d_in[3];
  const float*     value_b = (const float*)d_in[4];
  const float*     key_w   = (const float*)d_in[5];
  const float*     key_b   = (const float*)d_in[6];
  const float*     n1w     = (const float*)d_in[7];
  const float*     n2w     = (const float*)d_in[8];
  const float*     scw     = (const float*)d_in[9];
  const float*     cw      = (const float*)d_in[10];
  float* out = (float*)d_out;

  char* ws = (char*)d_ws;
  __bf16* embB  = (__bf16*)(ws);                         // 16 MiB
  __bf16* wtB   = (__bf16*)(ws + 16777216ULL);           // 20 MiB
  float*  proj  = (float*)(ws + 37748736ULL);            // 320 MiB
  float*  gateb = (float*)(ws + 373293056ULL);           // 128 KiB
  float*  rstdv = (float*)(ws + 373424128ULL);           // 128 KiB

  gather_k<<<(M_ * K_) / 256, 256, 0, stream>>>(tabs, hashids, embB);
  transpose_k<<<dim3(HID_ / 32, ENG_ / 32, HC_ + 1), dim3(32, 8), 0, stream>>>(
      key_w, value_w, wtB);
  gemm_bf16<<<dim3(N_ / 256, M_ / 128), 256, 0, stream>>>(embB, wtB, proj);
  gate_k<<<dim3(M_, HC_), 256, 0, stream>>>(proj, hidden, key_b, value_b,
                                            n1w, n2w, gateb, rstdv);
  out_k<<<(M_ * HID_) / 256, 256, 0, stream>>>(proj, value_b, gateb, rstdv,
                                               scw, cw, out);
}
